// SemanticHead_66597762891820
// MI455X (gfx1250) — compile-verified, matrix operand order checked by reading, not run on hardware
//
#include <hip/hip_runtime.h>
#include <stdint.h>

typedef __attribute__((ext_vector_type(16))) __bf16 v16bf;
typedef __attribute__((ext_vector_type(8)))  float  v8f;

union FragA { v16bf v; uint4 q[2]; unsigned short u[16]; };

__device__ __forceinline__ unsigned short f2bf(float f) {
  unsigned int u = __float_as_uint(f);
  u += 0x7FFFu + ((u >> 16) & 1u);          // round-to-nearest-even
  return (unsigned short)(u >> 16);
}
__device__ __forceinline__ float bf2f(unsigned short h) {
  return __uint_as_float(((unsigned int)h) << 16);
}
__device__ __forceinline__ float leaky(float v) { return v >= 0.f ? v : 0.1f * v; }

// ---------------------------------------------------------------------------
// Prep: convert p1 weights (fp32) into bf16 WMMA B-fragments in workspace.
// Layout per fragment: 32 lanes x 16 halves contiguous -> each lane loads its
// 32 bytes with two global_load_b128's.
// W1: 12 ktiles x 8 ntiles (49152 halves), W2: 4x4 (8192), W3: 2x2 (2048).
// ---------------------------------------------------------------------------
__global__ __launch_bounds__(256) void prep_weights(
    const float* __restrict__ W1, const float* __restrict__ W2,
    const float* __restrict__ W3, unsigned short* __restrict__ ws) {
  int t = blockIdx.x * 256 + threadIdx.x;
  if (t >= 59392) return;
  int half, lane, kt, nt, ldw;
  const float* W;
  if (t < 49152) {
    int u = t;
    half = u & 15; lane = (u >> 4) & 31; nt = (u >> 9) & 7; kt = u >> 12;
    W = W1; ldw = 128;
  } else if (t < 57344) {
    int u = t - 49152;
    half = u & 15; lane = (u >> 4) & 31; nt = (u >> 9) & 3; kt = u >> 11;
    W = W2; ldw = 64;
  } else {
    int u = t - 57344;
    half = u & 15; lane = (u >> 4) & 31; nt = (u >> 9) & 1; kt = u >> 10;
    W = W3; ldw = 32;
  }
  int k = ((lane >> 4) << 3) + (half & 7) + ((half & 8) ? 16 : 0);
  int n = (nt << 4) + (lane & 15);
  ws[t] = f2bf(W[(kt * 32 + k) * ldw + n]);
}

struct Params {
  const float* features;
  const float* lnw1; const float* lnb1; const float* b1;
  const float* lnw2; const float* lnb2; const float* b2;
  const float* lnw3; const float* lnb3; const float* b3;
  const float* lnw4; const float* lnb4; const float* W4; const float* b4;
  const float* q_lnw64; const float* q_lnb64; const float* qW1; const float* qb1;
  const float* q_lnw32; const float* q_lnb32; const float* qW2; const float* qb2;
  const float* q_lnw16; const float* q_lnb16; const float* qW3; const float* qb3;
  const unsigned short* wsbf;
  float* outp;   // [B,16]
  float* outb;   // [B,64]
};

#define XSTRIDE 392  // 384 + pad; multiple of 8 -> 16B-aligned b128 A loads

__global__ __launch_bounds__(128) void head_kernel(Params p) {
  __shared__ __align__(16) unsigned short Xn[64 * XSTRIDE];  // normalized bf16 tile
  __shared__ float Yb_all[4 * 16 * 128];                     // per-wave activation scratch
  __shared__ float ps[128], psq[128];
  __shared__ float meanA[64], rstdA[64], normsA[64];
  __shared__ float lnw2s[128], lnb2s[128];
  __shared__ float lnw3s[64], lnb3s[64];
  __shared__ float lnw4s[32], lnb4s[32];
  __shared__ float proj1row[64];

  const int tid  = threadIdx.x;
  const int b    = blockIdx.x;
  const int hw   = tid & 63;
  const int half = tid >> 6;

  // stage small LN params
  lnw2s[tid] = p.lnw2[tid]; lnb2s[tid] = p.lnb2[tid];
  if (tid < 64) { lnw3s[tid] = p.lnw3[tid]; lnb3s[tid] = p.lnb3[tid]; }
  if (tid < 32) { lnw4s[tid] = p.lnw4[tid]; lnb4s[tid] = p.lnb4[tid]; }

  // ---- Phase 1: single coalesced HBM pass; accumulate mean/var/sumsq; stash raw bf16
  const float* fb = p.features + (size_t)b * 24576;
  {
    float s = 0.f, q = 0.f;
    for (int i = 0; i < 192; ++i) {
      int c = half * 192 + i;
      float x = fb[c * 64 + hw];       // lanes sweep hw -> 128B/wave requests
      s += x; q += x * x;
      Xn[hw * XSTRIDE + c] = f2bf(x);
    }
    ps[tid] = s; psq[tid] = q;
  }
  __syncthreads();

  if (tid < 64) {
    float S = ps[tid] + ps[64 + tid];
    float Q = psq[tid] + psq[64 + tid];
    float mean = S * (1.f / 384.f);
    float var  = Q * (1.f / 384.f) - mean * mean;
    meanA[tid] = mean;
    rstdA[tid] = rsqrtf(var + 1e-5f);
    normsA[tid] = sqrtf(Q);            // channel L2 norm (saliency)
  }
  __syncthreads();

  // ---- Saliency binary mask: strict-rank top-16 (min-max norm is monotonic)
  if (tid < 64) {
    float me = normsA[tid];
    int cnt = 0;
    for (int j = 0; j < 64; ++j) cnt += (normsA[j] > me) ? 1 : 0;
    p.outb[(size_t)b * 64 + tid] = (cnt < 16) ? 1.f : 0.f;
  }

  // ---- Phase 2: in-LDS LayerNorm-1 (affine) -> normalized bf16 A-tile
  {
    float m = meanA[hw], r = rstdA[hw];
    for (int i = 0; i < 192; ++i) {
      int c = half * 192 + i;          // c uniform per wave -> scalar ln loads
      int idx = hw * XSTRIDE + c;
      float x = bf2f(Xn[idx]);
      Xn[idx] = f2bf((x - m) * r * p.lnw1[c] + p.lnb1[c]);
    }
  }
  __syncthreads();

  // ---- Phase 3: WMMA pipeline, one 16-pixel tile per wave -----------------
  const int lane = tid & 31;
  const int wv   = tid >> 5;
  const int g    = lane >> 4;
  const int mrow = lane & 15;
  const int hw0  = wv * 16;
  float* Yb = Yb_all + wv * (16 * 128);
  const uint4* w1f = (const uint4*)p.wsbf;
  const uint4* w2f = w1f + 6144;   // byte 98304
  const uint4* w3f = w1f + 7168;   // byte 114688

  // Layer 1: [16x384] @ [384x128]  (12 K-steps x 8 N-tiles)
  v8f acc[8];
#pragma unroll
  for (int nt = 0; nt < 8; ++nt) acc[nt] = {};
  for (int kt = 0; kt < 12; ++kt) {
    FragA A;
    const uint4* ap = (const uint4*)(Xn + ((hw0 + mrow) * XSTRIDE + kt * 32 + g * 8));
    A.q[0] = ap[0];          // K = g*8 + 0..7
    A.q[1] = ap[2];          // K = 16 + g*8 + 0..7
#pragma unroll
    for (int nt = 0; nt < 8; ++nt) {
      FragA Bf;
      const uint4* bp = w1f + ((kt * 8 + nt) * 64 + lane * 2);
      Bf.q[0] = bp[0]; Bf.q[1] = bp[1];
      acc[nt] = __builtin_amdgcn_wmma_f32_16x16x32_bf16(
          false, A.v, false, Bf.v, (short)0, acc[nt], false, false);
    }
  }
#pragma unroll
  for (int nt = 0; nt < 8; ++nt) {
    int n = nt * 16 + mrow;
    float bias = p.b1[n];
#pragma unroll
    for (int r = 0; r < 8; ++r)
      Yb[(g * 8 + r) * 128 + n] = leaky(acc[nt][r] + bias);
  }

  // LN over 128 (2 lanes per row, combine via shfl_xor 16)
  float mean1, rstd1;
  {
    float s = 0.f, q = 0.f;
    for (int j = 0; j < 64; ++j) {
      float v = Yb[mrow * 128 + g * 64 + j];
      s += v; q += v * v;
    }
    s += __shfl_xor(s, 16); q += __shfl_xor(q, 16);
    mean1 = s * (1.f / 128.f);
    rstd1 = rsqrtf(q * (1.f / 128.f) - mean1 * mean1 + 1e-5f);
  }

  // Layer 2: [16x128] @ [128x64]
  v8f acc2[4];
#pragma unroll
  for (int nt = 0; nt < 4; ++nt) acc2[nt] = {};
  for (int kt = 0; kt < 4; ++kt) {
    FragA A;
#pragma unroll
    for (int h = 0; h < 16; ++h) {
      int k = kt * 32 + g * 8 + (h & 7) + ((h & 8) ? 16 : 0);
      float y = Yb[mrow * 128 + k];
      A.u[h] = f2bf((y - mean1) * rstd1 * lnw2s[k] + lnb2s[k]);
    }
#pragma unroll
    for (int nt = 0; nt < 4; ++nt) {
      FragA Bf;
      const uint4* bp = w2f + ((kt * 4 + nt) * 64 + lane * 2);
      Bf.q[0] = bp[0]; Bf.q[1] = bp[1];
      acc2[nt] = __builtin_amdgcn_wmma_f32_16x16x32_bf16(
          false, A.v, false, Bf.v, (short)0, acc2[nt], false, false);
    }
  }
#pragma unroll
  for (int nt = 0; nt < 4; ++nt) {
    int n = nt * 16 + mrow;
    float bias = p.b2[n];
#pragma unroll
    for (int r = 0; r < 8; ++r)
      Yb[(g * 8 + r) * 64 + n] = leaky(acc2[nt][r] + bias);
  }

  float mean2, rstd2;
  {
    float s = 0.f, q = 0.f;
    for (int j = 0; j < 32; ++j) {
      float v = Yb[mrow * 64 + g * 32 + j];
      s += v; q += v * v;
    }
    s += __shfl_xor(s, 16); q += __shfl_xor(q, 16);
    mean2 = s * (1.f / 64.f);
    rstd2 = rsqrtf(q * (1.f / 64.f) - mean2 * mean2 + 1e-5f);
  }

  // Layer 3: [16x64] @ [64x32]
  v8f acc3[2];
#pragma unroll
  for (int nt = 0; nt < 2; ++nt) acc3[nt] = {};
  for (int kt = 0; kt < 2; ++kt) {
    FragA A;
#pragma unroll
    for (int h = 0; h < 16; ++h) {
      int k = kt * 32 + g * 8 + (h & 7) + ((h & 8) ? 16 : 0);
      float y = Yb[mrow * 64 + k];
      A.u[h] = f2bf((y - mean2) * rstd2 * lnw3s[k] + lnb3s[k]);
    }
#pragma unroll
    for (int nt = 0; nt < 2; ++nt) {
      FragA Bf;
      const uint4* bp = w3f + ((kt * 2 + nt) * 64 + lane * 2);
      Bf.q[0] = bp[0]; Bf.q[1] = bp[1];
      acc3[nt] = __builtin_amdgcn_wmma_f32_16x16x32_bf16(
          false, A.v, false, Bf.v, (short)0, acc3[nt], false, false);
    }
  }
#pragma unroll
  for (int nt = 0; nt < 2; ++nt) {
    int n = nt * 16 + mrow;
    float bias = p.b3[n];
#pragma unroll
    for (int r = 0; r < 8; ++r)
      Yb[(g * 8 + r) * 32 + n] = leaky(acc3[nt][r] + bias);
  }

  float mean3, rstd3;
  {
    float s = 0.f, q = 0.f;
    for (int j = 0; j < 16; ++j) {
      float v = Yb[mrow * 32 + g * 16 + j];
      s += v; q += v * v;
    }
    s += __shfl_xor(s, 16); q += __shfl_xor(q, 16);
    mean3 = s * (1.f / 32.f);
    rstd3 = rsqrtf(q * (1.f / 32.f) - mean3 * mean3 + 1e-5f);
  }

  // Layer 4 (32 -> 1) in VALU: 2 lanes per row, 16 MACs each
  {
    float s4 = 0.f;
    for (int j = 0; j < 16; ++j) {
      int c = g * 16 + j;
      float xn = (Yb[mrow * 32 + c] - mean3) * rstd3 * lnw4s[c] + lnb4s[c];
      s4 += xn * p.W4[c];
    }
    s4 += __shfl_xor(s4, 16);
    if (lane < 16) proj1row[hw0 + lane] = s4 + p.b4[0];
  }
  __syncthreads();

  // ---- Phase 4: p2 MLP on this image's 64-vector (wave 0, shuffles only) --
  if (tid < 32) {
    const int L = tid;
    float a0 = proj1row[L], a1 = proj1row[L + 32];
    float S = a0 + a1, Q = a0 * a0 + a1 * a1;
#pragma unroll
    for (int m = 16; m >= 1; m >>= 1) { S += __shfl_xor(S, m); Q += __shfl_xor(Q, m); }
    float mean = S * (1.f / 64.f);
    float rstd = rsqrtf(Q * (1.f / 64.f) - mean * mean + 1e-5f);

    float accv = p.qb1[L];
    for (int k = 0; k < 64; ++k) {
      float xn = (proj1row[k] - mean) * rstd * p.q_lnw64[k] + p.q_lnb64[k];
      accv += xn * p.qW1[k * 32 + L];
    }
    float h = leaky(accv);

    S = h; Q = h * h;
#pragma unroll
    for (int m = 16; m >= 1; m >>= 1) { S += __shfl_xor(S, m); Q += __shfl_xor(Q, m); }
    mean = S * (1.f / 32.f);
    rstd = rsqrtf(Q * (1.f / 32.f) - mean * mean + 1e-5f);
    float xn1 = (h - mean) * rstd * p.q_lnw32[L] + p.q_lnb32[L];

    const int n = L & 15;
    float acc2v = p.qb2[n];
    for (int k = 0; k < 32; ++k) acc2v += __shfl(xn1, k) * p.qW2[k * 16 + n];
    float h2 = leaky(acc2v);

    S = h2; Q = h2 * h2;                 // masks <=8 keep lanes 0..15 isolated
#pragma unroll
    for (int m = 8; m >= 1; m >>= 1) { S += __shfl_xor(S, m); Q += __shfl_xor(Q, m); }
    mean = S * (1.f / 16.f);
    rstd = rsqrtf(Q * (1.f / 16.f) - mean * mean + 1e-5f);
    float xn2 = (h2 - mean) * rstd * p.q_lnw16[n] + p.q_lnb16[n];

    float acc3v = p.qb3[n];
    for (int k = 0; k < 16; ++k) acc3v += __shfl(xn2, k) * p.qW3[k * 16 + n];
    if (L < 16) p.outp[(size_t)b * 16 + L] = acc3v;
  }
}

extern "C" void kernel_launch(void* const* d_in, const int* in_sizes, int n_in,
                              void* d_out, int out_size, void* d_ws, size_t ws_size,
                              hipStream_t stream) {
  (void)n_in; (void)out_size; (void)ws_size;
  const int Bn = in_sizes[0] / (384 * 64);
  unsigned short* ws = (unsigned short*)d_ws;   // needs 118,784 bytes

  prep_weights<<<(59392 + 255) / 256, 256, 0, stream>>>(
      (const float*)d_in[3], (const float*)d_in[7], (const float*)d_in[11], ws);

  Params p;
  p.features = (const float*)d_in[0];
  p.lnw1 = (const float*)d_in[1];  p.lnb1 = (const float*)d_in[2];  p.b1 = (const float*)d_in[4];
  p.lnw2 = (const float*)d_in[5];  p.lnb2 = (const float*)d_in[6];  p.b2 = (const float*)d_in[8];
  p.lnw3 = (const float*)d_in[9];  p.lnb3 = (const float*)d_in[10]; p.b3 = (const float*)d_in[12];
  p.lnw4 = (const float*)d_in[13]; p.lnb4 = (const float*)d_in[14];
  p.W4   = (const float*)d_in[15]; p.b4   = (const float*)d_in[16];
  p.q_lnw64 = (const float*)d_in[17]; p.q_lnb64 = (const float*)d_in[18];
  p.qW1     = (const float*)d_in[19]; p.qb1     = (const float*)d_in[20];
  p.q_lnw32 = (const float*)d_in[21]; p.q_lnb32 = (const float*)d_in[22];
  p.qW2     = (const float*)d_in[23]; p.qb2     = (const float*)d_in[24];
  p.q_lnw16 = (const float*)d_in[25]; p.q_lnb16 = (const float*)d_in[26];
  p.qW3     = (const float*)d_in[27]; p.qb3     = (const float*)d_in[28];
  p.wsbf = ws;
  p.outp = (float*)d_out;
  p.outb = (float*)d_out + (size_t)Bn * 16;

  head_kernel<<<Bn, 128, 0, stream>>>(p);
}
